// SeparableFCTP_21603685499504
// MI455X (gfx1250) — matrix operand order repaired and checked by reading, not compiled, not run on hardware
//
#include <hip/hip_runtime.h>
#include <hip/hip_bf16.h>

typedef __attribute__((ext_vector_type(16))) __bf16 bf16x16;
typedef __attribute__((ext_vector_type(8)))  float  f32x8;
typedef unsigned short u16;
typedef unsigned int   u32;
typedef __attribute__((ext_vector_type(4)))  u32    u32x4;

#define E_TOTAL   200000
#define NTILES    (E_TOTAL/16)     // 12500
#define WAVES     4
#define THREADS   (WAVES*32)
#define NGROUPS   (NTILES/WAVES)   // 3125
#define GRID      625

// ---- LDS layout (shared weight region, bf16, transposed: [N][K] K-contiguous) ----
#define W1T_OFF 0            // 64 x 64, stride 72
#define W1T_STR 72
#define W2T_OFF 4608         // 448 x 64, stride 72
#define W2T_STR 72
#define L0T_OFF 36864        // 192 x 192, stride 200
#define L0T_STR 200
#define L1T_OFF 75264        // 64 x 256, stride 264
#define L1T_STR 264
#define WEIGHT_B 184320      // bytes

// ---- per-wave scratch ----
#define AB_STR  264                       // A-tile bf16 stride (max K=256 + pad)
#define ABUF_B  (16*AB_STR*2)             // 8448
#define WB_STR  456                       // w-tile bf16 stride (448 + pad)
#define WBUF_B  (16*WB_STR*2)             // 14592
#define HB_STR  72                        // f32 h-tile stride (overlays w-tile)
#define GBUF_B  (16*64*4)                 // 4096
#define PW_B    (ABUF_B+WBUF_B+GBUF_B)    // 27136
#define SMEM_B  (WEIGHT_B + WAVES*PW_B)   // 292864 bytes < 320KB WGP LDS

#define SILU_CST 1.6765f   // 1/RMS(silu(z)), z~N(0,1)
#define SIG_CST  1.8461f   // 1/RMS(sigmoid(z))
#define INV_S3   0.57735026919f
#define INV_S2   0.70710678119f

__device__ __forceinline__ u16 f2bf(float f) {          // RNE f32->bf16
  u32 u = __builtin_bit_cast(u32, f);
  u32 r = u + 0x7FFFu + ((u >> 16) & 1u);
  return (u16)(r >> 16);
}
__device__ __forceinline__ u32 pack2(float lo, float hi) {
  return (u32)f2bf(lo) | ((u32)f2bf(hi) << 16);
}
__device__ __forceinline__ u32x4 pack8(const float* v) {
  u32x4 r;
  r.x = pack2(v[0], v[1]); r.y = pack2(v[2], v[3]);
  r.z = pack2(v[4], v[5]); r.w = pack2(v[6], v[7]);
  return r;
}
__device__ __forceinline__ float bf2f(u16 h) {
  u32 u = ((u32)h) << 16;
  return __builtin_bit_cast(float, u);
}
__device__ __forceinline__ float sigf(float x) { return 1.0f / (1.0f + __expf(-x)); }

union FragU { u32x4 q[2]; bf16x16 v; };

// A fragment: lane = (m = lane&15, kh = lane>>4), contiguous 16 K per half.
// VGPR0..7 hold K = k0+16*kh+{0..15}.
__device__ __forceinline__ bf16x16 lda(const u16* row, int k0, int kh) {
  const u32x4* p = (const u32x4*)(row + k0 + 16*kh);  // 16B aligned by layout
  FragU f;
  f.q[0] = p[0];        // K +0..7
  f.q[1] = p[1];        // K +8..15
  return f.v;
}
// B fragment: lane = (n = lane&15, kh = lane>>4), contiguous 16 K per half.
__device__ __forceinline__ bf16x16 ldb(const u16* row, int k0, int kh) {
  const u32x4* p = (const u32x4*)(row + k0 + 16*kh);  // 16B aligned by layout
  FragU f;
  f.q[0] = p[0];
  f.q[1] = p[1];
  return f.v;
}
#define WMMA_BF16(a,b,c) __builtin_amdgcn_wmma_f32_16x16x32_bf16(false,(a),false,(b),(short)0,(c),false,false)
#define FENCE() asm volatile("" ::: "memory")

__global__ void __launch_bounds__(THREADS) fctp_kernel(
    const float* __restrict__ node_input,  const float* __restrict__ edge_attr,
    const float* __restrict__ edge_scalars,
    const float* __restrict__ rad_w1,  const float* __restrict__ rad_b1,
    const float* __restrict__ rad_ln_g,const float* __restrict__ rad_ln_b,
    const float* __restrict__ rad_w2,  const float* __restrict__ rad_offset,
    const float* __restrict__ lin_w0,  const float* __restrict__ lin_b0,
    const float* __restrict__ lin_w1,  float* __restrict__ out)
{
  extern __shared__ char smem[];
  u16* ws   = (u16*)smem;
  u32* ws32 = (u32*)smem;
  const int tid  = threadIdx.x;
  const int wave = tid >> 5;
  const int lane = tid & 31;
  const int nl   = lane & 15;
  const int kh   = lane >> 4;

  // ---- stage all weights (f32 -> bf16, transposed) into shared LDS, once ----
  for (int i = tid; i < 64*32; i += THREADS) {           // rad_w1^T: 64x64
    int n = i >> 5, kp = i & 31, k = 2*kp;
    ws32[(W1T_OFF>>1) + n*(W1T_STR>>1) + kp] = pack2(rad_w1[k*64+n], rad_w1[(k+1)*64+n]);
  }
  for (int i = tid; i < 448*32; i += THREADS) {          // rad_w2^T: 448x64
    int n = i >> 5, kp = i & 31, k = 2*kp;
    ws32[(W2T_OFF>>1) + n*(W2T_STR>>1) + kp] = pack2(rad_w2[k*448+n], rad_w2[(k+1)*448+n]);
  }
  for (int i = tid; i < 192*96; i += THREADS) {          // lin_w0^T: 192x192
    int n = i / 96, kp = i % 96, k = 2*kp;
    ws32[(L0T_OFF>>1) + n*(L0T_STR>>1) + kp] = pack2(lin_w0[k*192+n], lin_w0[(k+1)*192+n]);
  }
  for (int i = tid; i < 64*128; i += THREADS) {          // lin_w1^T: 64x256
    int n = i >> 7, kp = i & 127, k = 2*kp;
    ws32[(L1T_OFF>>1) + n*(L1T_STR>>1) + kp] = pack2(lin_w1[k*64+n], lin_w1[(k+1)*64+n]);
  }
  __syncthreads();

  char*  pwv  = smem + WEIGHT_B + wave*PW_B;
  u16*   Abuf = (u16*)pwv;
  u16*   Wb   = (u16*)(pwv + ABUF_B);          // w tile, bf16 16x448
  float* Hb   = (float*)(pwv + ABUF_B);        // h tile, f32 16x64 (overlay)
  float* Gb   = (float*)(pwv + ABUF_B + WBUF_B);
  const u16* Arow = Abuf + nl*AB_STR;

  for (int grp = blockIdx.x; grp < NGROUPS; grp += (int)gridDim.x) {
    const int tile = grp*WAVES + wave;
    const int e0   = tile*16;

    if (grp + (int)gridDim.x < NGROUPS) {
      const int ne0 = (grp + (int)gridDim.x)*WAVES*16 + wave*16;
      __builtin_prefetch(edge_scalars + (size_t)ne0*64, 0, 1);  // global_prefetch_b8
    }

    // ---- stage edge_scalars tile -> A (bf16), b128 LDS stores ----
    {
      const float* src = edge_scalars + (size_t)(e0 + nl)*64 + kh*32;
      u32x4* dst = (u32x4*)(Abuf + nl*AB_STR + kh*32);
      #pragma unroll
      for (int g = 0; g < 4; ++g) {
        float4 a = *(const float4*)(src + 8*g);
        float4 b = *(const float4*)(src + 8*g + 4);
        u32x4 r;
        r.x = pack2(a.x, a.y); r.y = pack2(a.z, a.w);
        r.z = pack2(b.x, b.y); r.w = pack2(b.z, b.w);
        dst[g] = r;
      }
    }
    FENCE();

    // ---- GEMM1: h = scalars @ rad_w1 + b1  (16x64) ----
    #pragma unroll
    for (int nt = 0; nt < 4; ++nt) {
      const int n = nt*16 + nl;
      const u16* Brow = ws + W1T_OFF + n*W1T_STR;
      f32x8 acc = {};
      for (int k0 = 0; k0 < 64; k0 += 32)
        acc = WMMA_BF16(lda(Arow,k0,kh), ldb(Brow,k0,kh), acc);
      const float b = rad_b1[n];
      #pragma unroll
      for (int r = 0; r < 8; ++r)
        Hb[(r + 8*kh)*HB_STR + n] = acc[r] + b;
    }
    FENCE();

    // ---- LayerNorm + SiLU -> A (bf16) ----
    {
      const float* hrow = Hb + nl*HB_STR + kh*32;
      float sum = 0.f, ssq = 0.f;
      #pragma unroll
      for (int c = 0; c < 32; ++c) { float x = hrow[c]; sum += x; ssq += x*x; }
      sum += __shfl_xor(sum, 16, 32);
      ssq += __shfl_xor(ssq, 16, 32);
      const float mu  = sum * (1.f/64.f);
      const float var = ssq * (1.f/64.f) - mu*mu;
      const float rs  = rsqrtf(var + 1e-5f);
      u32x4* ar4 = (u32x4*)(Abuf + nl*AB_STR + kh*32);
      #pragma unroll
      for (int g = 0; g < 4; ++g) {
        float vv[8];
        #pragma unroll
        for (int t = 0; t < 8; ++t) {
          int col = kh*32 + g*8 + t;
          float x = (hrow[g*8 + t] - mu)*rs*rad_ln_g[col] + rad_ln_b[col];
          vv[t] = x * sigf(x);
        }
        ar4[g] = pack8(vv);
      }
    }
    FENCE();

    // ---- GEMM2: w = h' @ rad_w2 + offset  (16x448, stored bf16) ----
    for (int nt = 0; nt < 28; ++nt) {
      const int n = nt*16 + nl;
      const u16* Brow = ws + W2T_OFF + n*W2T_STR;
      f32x8 acc = {};
      for (int k0 = 0; k0 < 64; k0 += 32)
        acc = WMMA_BF16(lda(Arow,k0,kh), ldb(Brow,k0,kh), acc);
      const float b = rad_offset[n];
      #pragma unroll
      for (int r = 0; r < 8; ++r)
        Wb[(r + 8*kh)*WB_STR + n] = f2bf(acc[r] + b);
    }
    FENCE();

    // ---- per-edge tensor-product operands ----
    const int    erow = e0 + nl;
    const float* xr   = node_input + (size_t)erow*320;
    const float4 sh   = *(const float4*)(edge_attr + (size_t)erow*4);
    const float  sh0 = sh.x;
    const float  sh1v[3] = { sh.y, sh.z, sh.w };
    const u16*   wrow = Wb + nl*WB_STR;

    // ---- build scal = [p0(128) | p3(64)] -> A ----
    {
      u32x4* ar4 = (u32x4*)(Abuf + nl*AB_STR);   // 8 bf16 cols per u32x4
      for (int g = kh*12; g < kh*12 + 12; ++g) {
        float vv[8];
        #pragma unroll
        for (int t = 0; t < 8; ++t) {
          int n = g*8 + t;
          if (n < 128) {
            vv[t] = bf2f(wrow[n]) * xr[n] * sh0;
          } else {
            int j = n - 128;
            const float* x1 = xr + 128 + 3*j;
            float d = x1[0]*sh1v[0] + x1[1]*sh1v[1] + x1[2]*sh1v[2];
            vv[t] = bf2f(wrow[320 + j]) * d * INV_S3;
          }
        }
        ar4[g] = pack8(vv);
      }
    }
    FENCE();

    // ---- GEMM3: s = scal @ lin_w0 + b0 ; silu/sigmoid split ----
    for (int nt = 0; nt < 12; ++nt) {
      const int n = nt*16 + nl;
      const u16* Brow = ws + L0T_OFF + n*L0T_STR;
      f32x8 acc = {};
      for (int k0 = 0; k0 < 192; k0 += 32)
        acc = WMMA_BF16(lda(Arow,k0,kh), ldb(Brow,k0,kh), acc);
      const float b = lin_b0[n];
      #pragma unroll
      for (int r = 0; r < 8; ++r) {
        const int m = r + 8*kh;
        const float sv = acc[r] + b;
        if (n < 128) out[(size_t)(e0+m)*320 + n] = SILU_CST * sv * sigf(sv);
        else         Gb[m*64 + (n-128)] = SIG_CST * sigf(sv);
      }
    }
    FENCE();

    // ---- GEMM4 (x3 components): v_c = vec_c @ lin_w1 ; gate ; store ----
    #pragma unroll
    for (int c = 0; c < 3; ++c) {
      const int c1 = (c+1)%3, c2i = (c+2)%3;
      u32x4* ar4 = (u32x4*)(Abuf + nl*AB_STR);
      for (int g = kh*16; g < kh*16 + 16; ++g) {
        float vv[8];
        #pragma unroll
        for (int t = 0; t < 8; ++t) {
          int u = g*8 + t;
          if (u < 128) {
            vv[t] = bf2f(wrow[128 + u]) * xr[u] * sh1v[c];
          } else if (u < 192) {
            int j = u - 128;
            vv[t] = bf2f(wrow[256 + j]) * xr[128 + 3*j + c] * sh0;
          } else {
            int j = u - 192;
            const float* x1 = xr + 128 + 3*j;
            float cr = x1[c1]*sh1v[c2i] - x1[c2i]*sh1v[c1];
            vv[t] = bf2f(wrow[384 + j]) * cr * INV_S2;
          }
        }
        ar4[g] = pack8(vv);
      }
      FENCE();
      for (int nt = 0; nt < 4; ++nt) {
        const int wc = nt*16 + nl;
        const u16* Brow = ws + L1T_OFF + wc*L1T_STR;
        f32x8 acc = {};
        for (int k0 = 0; k0 < 256; k0 += 32)
          acc = WMMA_BF16(lda(Arow,k0,kh), ldb(Brow,k0,kh), acc);
        #pragma unroll
        for (int r = 0; r < 8; ++r) {
          const int m = r + 8*kh;
          out[(size_t)(e0+m)*320 + 128 + wc*3 + c] = Gb[m*64 + wc] * acc[r];
        }
      }
      FENCE();
    }
  }
}

extern "C" void kernel_launch(void* const* d_in, const int* in_sizes, int n_in,
                              void* d_out, int out_size, void* d_ws, size_t ws_size,
                              hipStream_t stream) {
  (void)in_sizes; (void)n_in; (void)out_size; (void)d_ws; (void)ws_size;
  const float* node_input   = (const float*)d_in[0];
  const float* edge_attr    = (const float*)d_in[1];
  const float* edge_scalars = (const float*)d_in[2];
  const float* rad_w1   = (const float*)d_in[3];
  const float* rad_b1   = (const float*)d_in[4];
  const float* rad_ln_g = (const float*)d_in[5];
  const float* rad_ln_b = (const float*)d_in[6];
  const float* rad_w2   = (const float*)d_in[7];
  const float* rad_off  = (const float*)d_in[8];
  const float* lin_w0   = (const float*)d_in[9];
  const float* lin_b0   = (const float*)d_in[10];
  const float* lin_w1   = (const float*)d_in[11];
  float* out = (float*)d_out;

  fctp_kernel<<<dim3(GRID), dim3(THREADS), SMEM_B, stream>>>(
      node_input, edge_attr, edge_scalars, rad_w1, rad_b1, rad_ln_g, rad_ln_b,
      rad_w2, rad_off, lin_w0, lin_b0, lin_w1, out);
}
